// MultiheadAttention_18940805776191
// MI455X (gfx1250) — matrix operand order repaired and checked by reading, not compiled, not run on hardware
//
#include <hip/hip_runtime.h>
#include <hip/hip_bf16.h>
#include <math.h>

#define BB 8
#define LL 1024
#define EE 1024
#define HH 16
#define DD 64
#define NROW (BB * LL)   // 8192

typedef __attribute__((ext_vector_type(16))) __bf16 bf16x16;
typedef __attribute__((ext_vector_type(8)))  float  f32x8;
typedef __attribute__((ext_vector_type(4)))  unsigned int u32x4;
typedef __attribute__((ext_vector_type(8)))  int    i32x8;
typedef __attribute__((ext_vector_type(4)))  int    i32x4;

union FragAB { bf16x16 v; float4 f4[2]; unsigned short us[16]; };
union FragC  { f32x8   v; float  f[8]; };
union U32h   { float4 f4[4]; unsigned short us[32]; };

__device__ __forceinline__ unsigned short f2bf(float f) {
  union { float f; unsigned int u; } v; v.f = f;
  unsigned int r = v.u + 0x7fffu + ((v.u >> 16) & 1u);
  return (unsigned short)(r >> 16);
}

// ---------------------------------------------------------------------------
// Tensor Data Mover: 2D tile (rows x rowBytes) from global (strideBytes row
// pitch) into contiguous LDS at ldsOff.  data_size = 8-byte units.
// D# layout per CDNA5 ISA:
//   g0 = { count=1, lds_addr, global_addr[31:0], ga[56:32] | type=2<<30 }
//   g1 = { dsz<<16, dim0<<16, dim1<<16... } packed as below; mask=0 (no cluster)
// ---------------------------------------------------------------------------
__device__ __forceinline__ void tdm_load_tile(unsigned ldsOff, const void* gp,
                                              unsigned rows, unsigned rowBytes,
                                              unsigned strideBytes) {
  unsigned long long ga = (unsigned long long)gp;
  unsigned du0 = rowBytes    >> 3;   // tile_dim0 / tensor_dim0 in 8B units
  unsigned su0 = strideBytes >> 3;   // tensor_dim0_stride in 8B units
  u32x4 g0 = { 1u, ldsOff, (unsigned)ga,
               (unsigned)((ga >> 32) & 0x01ffffffu) | 0x80000000u };
  i32x8 g1 = { (int)(3u << 16),      // [17:16] data_size = 3 (8 bytes)
               (int)(du0 << 16),     // [79:48] tensor_dim0 -> low 16 here
               (int)(rows << 16),    // tensor_dim0 hi | [111:80] tensor_dim1 lo
               (int)(du0 << 16),     // tensor_dim1 hi | [127:112] tile_dim0
               (int)rows,            // [143:128] tile_dim1 | tile_dim2 = 0
               (int)su0,             // [207:160] tensor_dim0_stride lo32
               0, 0 };
  i32x4 z4 = { 0, 0, 0, 0 };
#if defined(__clang_major__) && (__clang_major__ >= 23)
  i32x8 z8 = { 0, 0, 0, 0, 0, 0, 0, 0 };
  __builtin_amdgcn_tensor_load_to_lds(g0, g1, z4, z4, z8, 0);
#else
  __builtin_amdgcn_tensor_load_to_lds(g0, g1, z4, z4, 0);
#endif
}

__device__ __forceinline__ unsigned lds_off(const void* p) {
  return (unsigned)(unsigned long long)p;   // low 32 bits of flat addr = LDS offset
}

// ---------------------------------------------------------------------------
// K1: LayerNorm q/k/v -> bf16.  One wave per row, 4 rows per 128-thread block.
// ---------------------------------------------------------------------------
__global__ void __launch_bounds__(128) ln_kernel(
    const float* __restrict__ q, const float* __restrict__ k,
    const float* __restrict__ v, const float* __restrict__ g,
    const float* __restrict__ be, unsigned short* __restrict__ xn) {
  int row  = blockIdx.x * 4 + (threadIdx.x >> 5);
  int lane = threadIdx.x & 31;
  int t    = row >> 13;            // row / 8192 : which tensor
  int r    = row & (NROW - 1);
  const float* src = (t == 0) ? q : ((t == 1) ? k : v);
  const float* x = src + (size_t)r * EE;
  float xv[32], s = 0.f, ss = 0.f;
#pragma unroll
  for (int j = 0; j < 32; ++j) {
    float f = x[j * 32 + lane];
    xv[j] = f; s += f; ss += f * f;
  }
#pragma unroll
  for (int off = 16; off > 0; off >>= 1) {
    s  += __shfl_xor(s,  off, 32);
    ss += __shfl_xor(ss, off, 32);
  }
  float mean = s * (1.f / EE);
  float var  = ss * (1.f / EE) - mean * mean;
  float inv  = rsqrtf(var + 1e-5f);
  unsigned short* dst = xn + (size_t)row * EE;
#pragma unroll
  for (int j = 0; j < 32; ++j) {
    int e = j * 32 + lane;
    dst[e] = f2bf((xv[j] - mean) * inv * g[e] + be[e]);
  }
}

// ---------------------------------------------------------------------------
// K2: weight f32 -> bf16
// ---------------------------------------------------------------------------
__global__ void cvtw_kernel(const float* __restrict__ wi,
                            const float* __restrict__ wo,
                            unsigned short* __restrict__ wib,
                            unsigned short* __restrict__ wob) {
  const int NI = 3 * EE * EE, NO = EE * EE;
  for (int i = blockIdx.x * blockDim.x + threadIdx.x; i < NI + NO;
       i += gridDim.x * blockDim.x) {
    if (i < NI) wib[i] = f2bf(wi[i]);
    else        wob[i - NI] = f2bf(wo[i - NI]);
  }
}

// ---------------------------------------------------------------------------
// K3: in-proj GEMM  out[n,o] = sum_e xn[n,e]*W[o,e] + b[o]  (per q/k/v)
// 64x64 block tile, 4 waves.  TDM-staged, double-buffered LDS tiles:
// wave0 issues tensor_load_to_lds for step k+1 while all waves run WMMA on
// step k; s_wait_tensorcnt + workgroup barrier gate buffer reuse.
// Output stored bf16 in [B,H,L,D] layout.
// ---------------------------------------------------------------------------
__global__ void __launch_bounds__(128) gemm_qkv_kernel(
    const unsigned short* __restrict__ xn,
    const unsigned short* __restrict__ wib,
    const float* __restrict__ bias,
    unsigned short* __restrict__ qkv) {
  __shared__ unsigned short As[2][64 * 32];
  __shared__ unsigned short Bs[2][64 * 32];
  int which = blockIdx.z;
  int o0 = blockIdx.x * 64;
  int n0 = blockIdx.y * 64;
  const unsigned short* A = xn  + (size_t)which * NROW * EE + (size_t)n0 * EE;
  const unsigned short* W = wib + (size_t)which * EE * EE   + (size_t)o0 * EE;
  const float* bb = bias + which * EE;

  int t = threadIdx.x;
  int wave = t >> 5, lane = t & 31, m = lane & 15, hi = lane >> 4;

  FragC acc[4];
  for (int i = 0; i < 4; ++i)
    for (int r = 0; r < 8; ++r) acc[i].f[r] = 0.f;

  if (t < 32) {
    tdm_load_tile(lds_off(&As[0][0]), A, 64, 64, EE * 2);
    tdm_load_tile(lds_off(&Bs[0][0]), W, 64, 64, EE * 2);
  }
  int arow = wave * 16 + m;
  for (int ks = 0; ks < 32; ++ks) {
    if (t < 32) __builtin_amdgcn_s_wait_tensorcnt(0);
    __syncthreads();
    if (t < 32 && ks + 1 < 32) {
      tdm_load_tile(lds_off(&As[(ks + 1) & 1][0]), A + (ks + 1) * 32, 64, 64, EE * 2);
      tdm_load_tile(lds_off(&Bs[(ks + 1) & 1][0]), W + (ks + 1) * 32, 64, 64, EE * 2);
    }
    const unsigned short* as = As[ks & 1];
    const unsigned short* bs = Bs[ks & 1];
    FragAB a;
    a.f4[0] = *(const float4*)&as[arow * 32 + (hi ? 8 : 0)];
    a.f4[1] = *(const float4*)&as[arow * 32 + (hi ? 24 : 16)];
#pragma unroll
    for (int nt = 0; nt < 4; ++nt) {
      FragAB bf;
      int brow = nt * 16 + m;
      bf.f4[0] = *(const float4*)&bs[brow * 32 + (hi ? 8 : 0)];
      bf.f4[1] = *(const float4*)&bs[brow * 32 + (hi ? 24 : 16)];
      acc[nt].v = __builtin_amdgcn_wmma_f32_16x16x32_bf16(
          false, a.v, false, bf.v, (short)0, acc[nt].v, false, false);
    }
  }
#pragma unroll
  for (int nt = 0; nt < 4; ++nt) {
    int ol = o0 + nt * 16 + m;
    int hh = ol >> 6, dd = ol & 63;
    float bv = bb[ol];
#pragma unroll
    for (int r = 0; r < 8; ++r) {
      int n = n0 + wave * 16 + r + hi * 8;
      int bi = n >> 10, li = n & 1023;
      qkv[(size_t)which * ((size_t)NROW * EE) +
          ((size_t)(bi * HH + hh) * LL + li) * DD + dd] = f2bf(acc[nt].f[r] + bv);
    }
  }
}

// ---------------------------------------------------------------------------
// K4: attention.  Block = (16-query-row tile, batch).  64 threads (2 waves).
// Loops over 16 heads: q tile via TDM -> scores (WMMA) -> softmax (f32 LDS)
// -> heatmap accum (global, block-owned rows) -> probs bf16 -> ctx (WMMA with
// transposed v stage).  ctx written bf16 in [B,L,E].
// ---------------------------------------------------------------------------
__global__ void __launch_bounds__(64) attn_kernel(
    const unsigned short* __restrict__ qkv,
    unsigned short* __restrict__ ctx,
    float* __restrict__ heat) {
  __shared__ unsigned short qs[16 * 64];     //  2 KB
  __shared__ float          sc[16 * 1024];   // 64 KB scores (f32)
  __shared__ unsigned short pb[16 * 1024];   // 32 KB probs (bf16)
  __shared__ unsigned short vT[64 * 32];     //  4 KB transposed v stage
  __shared__ float          red[64];
  __shared__ float          rowstat[16];

  int qt = blockIdx.x, b = blockIdx.y;
  int q0 = qt * 16;
  int t = threadIdx.x, wave = t >> 5, lane = t & 31, m = lane & 15, hi = lane >> 4;

  const unsigned short* qg = qkv;
  const unsigned short* kg = qkv + (size_t)NROW * EE;
  const unsigned short* vg = qkv + 2 * (size_t)NROW * EE;

  for (int h = 0; h < HH; ++h) {
    size_t hb = (size_t)(b * HH + h) * LL * DD;
    // q tile (16 x 64 bf16, fully contiguous 2 KB) -> LDS via TDM
    if (t < 32) {
      tdm_load_tile(lds_off(&qs[0]), &qg[hb + (size_t)q0 * DD], 1, 16 * DD * 2, 16 * DD * 2);
      __builtin_amdgcn_s_wait_tensorcnt(0);
    }
    __syncthreads();
    // ---- scores: 16 x 1024, waves split the 64 N-tiles ----
    for (int nt = wave; nt < 64; nt += 2) {
      FragC acc;
      for (int r = 0; r < 8; ++r) acc.f[r] = 0.f;
#pragma unroll
      for (int ds = 0; ds < 2; ++ds) {
        FragAB a, bf;
        a.f4[0] = *(const float4*)&qs[m * 64 + ds * 32 + (hi ? 8 : 0)];
        a.f4[1] = *(const float4*)&qs[m * 64 + ds * 32 + (hi ? 24 : 16)];
        const unsigned short* kp =
            &kg[hb + (size_t)(nt * 16 + m) * DD + ds * 32 + (hi ? 8 : 0)];
        bf.f4[0] = *(const float4*)kp;
        bf.f4[1] = *(const float4*)(kp + 16);
        acc.v = __builtin_amdgcn_wmma_f32_16x16x32_bf16(
            false, a.v, false, bf.v, (short)0, acc.v, false, false);
      }
#pragma unroll
      for (int r = 0; r < 8; ++r)
        sc[(r + hi * 8) * 1024 + nt * 16 + m] = acc.f[r] * 0.125f;
    }
    __syncthreads();
    // ---- softmax over 1024 keys; thread t owns row t/4, quarter t%4 ----
    int row = t >> 2, seg = t & 3;
    float* sr = &sc[row * 1024 + seg * 256];
    float mx = -3.4e38f;
#pragma unroll 4
    for (int j = 0; j < 256; ++j) mx = fmaxf(mx, sr[j]);
    red[t] = mx;
    __syncthreads();
    if (t < 16)
      rowstat[t] = fmaxf(fmaxf(red[t * 4], red[t * 4 + 1]),
                         fmaxf(red[t * 4 + 2], red[t * 4 + 3]));
    __syncthreads();
    float rm = rowstat[row], sum = 0.f;
#pragma unroll 4
    for (int j = 0; j < 256; ++j) {
      float e = __expf(sr[j] - rm);
      sr[j] = e; sum += e;
    }
    red[t] = sum;
    __syncthreads();
    if (t < 16)
      rowstat[t] = 1.f / (red[t * 4] + red[t * 4 + 1] + red[t * 4 + 2] + red[t * 4 + 3]);
    __syncthreads();
    float inv = rowstat[row];
    unsigned short* pr = &pb[row * 1024 + seg * 256];
    float* hr = &heat[(size_t)(b * LL + q0 + row) * LL + seg * 256];
#pragma unroll 4
    for (int j = 0; j < 256; ++j) {
      float p = sr[j] * inv;
      pr[j] = f2bf(p);
      float hv = p * (1.f / HH);
      hr[j] = (h == 0) ? hv : (hr[j] + hv);
    }
    __syncthreads();
    // ---- ctx = probs @ v : 16 x 64, K = 1024 in steps of 32 ----
    FragC cacc[2];
    for (int i = 0; i < 2; ++i)
      for (int r = 0; r < 8; ++r) cacc[i].f[r] = 0.f;
    for (int ks = 0; ks < 32; ++ks) {
      { // stage v transposed: vT[d][key]
        int kk = t >> 1, dh = (t & 1) * 32;
        U32h tmp;
        const unsigned short* vp = &vg[hb + (size_t)(ks * 32 + kk) * DD + dh];
        tmp.f4[0] = *(const float4*)vp;
        tmp.f4[1] = *(const float4*)(vp + 8);
        tmp.f4[2] = *(const float4*)(vp + 16);
        tmp.f4[3] = *(const float4*)(vp + 24);
#pragma unroll
        for (int x = 0; x < 32; ++x) vT[(dh + x) * 32 + kk] = tmp.us[x];
      }
      __syncthreads();
      FragAB a;
      a.f4[0] = *(const float4*)&pb[m * 1024 + ks * 32 + (hi ? 8 : 0)];
      a.f4[1] = *(const float4*)&pb[m * 1024 + ks * 32 + (hi ? 24 : 16)];
#pragma unroll
      for (int ti = 0; ti < 2; ++ti) {
        int nt = wave + ti * 2;
        FragAB bf;
        bf.f4[0] = *(const float4*)&vT[(nt * 16 + m) * 32 + (hi ? 8 : 0)];
        bf.f4[1] = *(const float4*)&vT[(nt * 16 + m) * 32 + (hi ? 24 : 16)];
        cacc[ti].v = __builtin_amdgcn_wmma_f32_16x16x32_bf16(
            false, a.v, false, bf.v, (short)0, cacc[ti].v, false, false);
      }
      __syncthreads();
    }
#pragma unroll
    for (int ti = 0; ti < 2; ++ti) {
      int o = h * 64 + (wave + ti * 2) * 16 + m;
#pragma unroll
      for (int r = 0; r < 8; ++r) {
        int qm = q0 + r + hi * 8;
        ctx[(size_t)(b * LL + qm) * EE + o] = f2bf(cacc[ti].f[r]);
      }
    }
    __syncthreads();
  }
}

// ---------------------------------------------------------------------------
// K5: out-proj GEMM + bias + residual -> f32 d_out  (TDM-staged like K3)
// ---------------------------------------------------------------------------
__global__ void __launch_bounds__(128) gemm_out_kernel(
    const unsigned short* __restrict__ ctx,
    const unsigned short* __restrict__ wob,
    const float* __restrict__ bo,
    const float* __restrict__ query,
    float* __restrict__ out) {
  __shared__ unsigned short As[2][64 * 32];
  __shared__ unsigned short Bs[2][64 * 32];
  int o0 = blockIdx.x * 64;
  int n0 = blockIdx.y * 64;
  const unsigned short* A = ctx + (size_t)n0 * EE;
  const unsigned short* W = wob + (size_t)o0 * EE;

  int t = threadIdx.x;
  int wave = t >> 5, lane = t & 31, m = lane & 15, hi = lane >> 4;

  FragC acc[4];
  for (int i = 0; i < 4; ++i)
    for (int r = 0; r < 8; ++r) acc[i].f[r] = 0.f;

  if (t < 32) {
    tdm_load_tile(lds_off(&As[0][0]), A, 64, 64, EE * 2);
    tdm_load_tile(lds_off(&Bs[0][0]), W, 64, 64, EE * 2);
  }
  int arow = wave * 16 + m;
  for (int ks = 0; ks < 32; ++ks) {
    if (t < 32) __builtin_amdgcn_s_wait_tensorcnt(0);
    __syncthreads();
    if (t < 32 && ks + 1 < 32) {
      tdm_load_tile(lds_off(&As[(ks + 1) & 1][0]), A + (ks + 1) * 32, 64, 64, EE * 2);
      tdm_load_tile(lds_off(&Bs[(ks + 1) & 1][0]), W + (ks + 1) * 32, 64, 64, EE * 2);
    }
    const unsigned short* as = As[ks & 1];
    const unsigned short* bs = Bs[ks & 1];
    FragAB a;
    a.f4[0] = *(const float4*)&as[arow * 32 + (hi ? 8 : 0)];
    a.f4[1] = *(const float4*)&as[arow * 32 + (hi ? 24 : 16)];
#pragma unroll
    for (int nt = 0; nt < 4; ++nt) {
      FragAB bf;
      int brow = nt * 16 + m;
      bf.f4[0] = *(const float4*)&bs[brow * 32 + (hi ? 8 : 0)];
      bf.f4[1] = *(const float4*)&bs[brow * 32 + (hi ? 24 : 16)];
      acc[nt].v = __builtin_amdgcn_wmma_f32_16x16x32_bf16(
          false, a.v, false, bf.v, (short)0, acc[nt].v, false, false);
    }
  }
#pragma unroll
  for (int nt = 0; nt < 4; ++nt) {
    int ol = o0 + nt * 16 + m;
    float bv = bo[ol];
#pragma unroll
    for (int r = 0; r < 8; ++r) {
      int n = n0 + wave * 16 + r + hi * 8;
      out[(size_t)n * EE + ol] = acc[nt].f[r] + bv + query[(size_t)n * EE + ol];
    }
  }
}

// ---------------------------------------------------------------------------
extern "C" void kernel_launch(void* const* d_in, const int* in_sizes, int n_in,
                              void* d_out, int out_size, void* d_ws, size_t ws_size,
                              hipStream_t stream) {
  (void)in_sizes; (void)n_in; (void)out_size; (void)ws_size;
  const float* query = (const float*)d_in[0];
  const float* key   = (const float*)d_in[1];
  const float* value = (const float*)d_in[2];
  const float* gamma = (const float*)d_in[3];
  const float* beta  = (const float*)d_in[4];
  const float* wi    = (const float*)d_in[5];
  const float* bi    = (const float*)d_in[6];
  const float* wo    = (const float*)d_in[7];
  const float* bo    = (const float*)d_in[8];

  char* ws = (char*)d_ws;
  size_t off = 0;
  unsigned short* xn  = (unsigned short*)(ws + off); off += (size_t)3 * NROW * EE * 2;  // 48 MB
  unsigned short* wib = (unsigned short*)(ws + off); off += (size_t)3 * EE * EE * 2;    //  6 MB
  unsigned short* wob = (unsigned short*)(ws + off); off += (size_t)EE * EE * 2;        //  2 MB
  unsigned short* qkv = (unsigned short*)(ws + off); off += (size_t)3 * NROW * EE * 2;  // 48 MB
  unsigned short* ctx = (unsigned short*)(ws + off); off += (size_t)NROW * EE * 2;      // 16 MB

  float* out  = (float*)d_out;
  float* heat = out + (size_t)NROW * EE;

  ln_kernel<<<(3 * NROW) / 4, 128, 0, stream>>>(query, key, value, gamma, beta, xn);
  cvtw_kernel<<<4096, 256, 0, stream>>>(wi, wo, wib, wob);
  gemm_qkv_kernel<<<dim3(EE / 64, NROW / 64, 3), 128, 0, stream>>>(xn, wib, bi, qkv);
  attn_kernel<<<dim3(LL / 16, BB), 64, 0, stream>>>(qkv, ctx, heat);
  gemm_out_kernel<<<dim3(EE / 64, NROW / 64), 128, 0, stream>>>(ctx, wob, bo, query, out);
}
